// GINEEncoder_2422361554984
// MI455X (gfx1250) — matrix operand order repaired and checked by reading, not compiled, not run on hardware
//
#include <hip/hip_runtime.h>

#define NN 50000
#define EE 800000
#define FN 128
#define FE 16
#define HH 256
#define H2 512
#define LL 4
#define GG 64

typedef __attribute__((ext_vector_type(16))) __bf16 v16bf;
typedef __attribute__((ext_vector_type(8)))  __bf16 v8bf;
typedef __attribute__((ext_vector_type(8)))  float  v8f;
typedef __attribute__((ext_vector_type(4)))  unsigned v4u;
typedef __attribute__((ext_vector_type(8)))  int      v8i;
typedef __attribute__((ext_vector_type(4)))  int      v4i;

#if defined(__has_builtin)
#if __has_builtin(__builtin_amdgcn_tensor_load_to_lds) && __has_builtin(__builtin_amdgcn_s_wait_tensorcnt)
#define USE_TDM 1
#endif
#endif
#ifndef USE_TDM
#define USE_TDM 0
#endif

#if USE_TDM
#pragma message("CDNA5: TDM tensor_load_to_lds staging ENABLED")
#else
#pragma message("CDNA5: TDM builtin not found, using b128 copy staging")
#endif

#define LDS_STRIDE 40                    // bf16/row: 80B = 20 banks -> conflict-free b128 walk
#define LDS_CHUNK  (64 * LDS_STRIDE)     // one 32(k) x 64(n) staged tile, padded rows (elements)
#define LDS_BYTES  (2 * LDS_CHUNK * 2)   // double buffered

__device__ __forceinline__ float san(float f) { return f == f ? f : 0.0f; }

// B fragment: lane reads two 8-bf16 runs (k = 8h..8h+7 and 16+8h..16+8h+7) from LDS tile cur[n][LDS_STRIDE].
__device__ __forceinline__ v16bf load_b_frag(const __bf16* cur, int lane, int sub) {
  int n = (sub << 4) + (lane & 15);
  int kofs = (lane >> 4) << 3;
  v8bf lo = *(const v8bf*)(cur + n * LDS_STRIDE + kofs);
  v8bf hi = *(const v8bf*)(cur + n * LDS_STRIDE + kofs + 16);
  return __builtin_shufflevector(lo, hi, 0, 1, 2, 3, 4, 5, 6, 7, 8, 9, 10, 11, 12, 13, 14, 15);
}

// Fallback staging: 32(k) x 64(n) chunk of pre-transposed bf16 weights Wt[n][Kp] via b128 copies.
__device__ __forceinline__ void stage_copy(__bf16* buf, const __bf16* __restrict__ Wt, int Kp,
                                           int k0, int n0, int tid) {
  int n = tid >> 2, q = (tid & 3) << 3;
  *(v8bf*)(buf + n * LDS_STRIDE + q) = *(const v8bf*)(Wt + (size_t)(n0 + n) * Kp + k0 + q);
}

// Issue staging of one chunk (TDM: async DMA, no wait; fallback: direct copy).
// ldsoff = byte offset of the destination buffer within the dynamic-LDS allocation (base 0).
__device__ __forceinline__ void stage_issue(__bf16* buf, const __bf16* __restrict__ Wt, int Kp,
                                            int k0, int n0, int tid, unsigned ldsoff) {
#if USE_TDM
  if (tid < 32) {  // one TDM issue per block (TDM ignores EXEC; keep it to wave 0)
    unsigned long long ga = (unsigned long long)(size_t)(const void*)(Wt + (size_t)n0 * Kp + k0);
    v4u g0 = { 1u,                                   // count=1 valid descriptor
               ldsoff,                               // LDS destination address
               (unsigned)ga,
               (unsigned)((ga >> 32) & 0x1FFFFFFull) | (2u << 30) };  // addr[56:32] | type=2
    v8i g1 = { (int)0x06D10000u,   // data_size=2B | pad_enable | pad_interval=16dw | pad_amount=4dw
               (int)(32u << 16),   // tensor_dim0 = 32
               (int)(64u << 16),   // tensor_dim1 = 64
               (int)(32u << 16),   // tile_dim0   = 32
               64,                 // tile_dim1   = 64
               Kp,                 // tensor_dim0_stride (data_size units)
               0, 0 };
    v4i z4 = { 0, 0, 0, 0 };
#if defined(__clang_major__) && __clang_major__ >= 23
    v8i z8 = { 0, 0, 0, 0, 0, 0, 0, 0 };
    __builtin_amdgcn_tensor_load_to_lds(g0, g1, z4, z4, z8, 0);
#else
    __builtin_amdgcn_tensor_load_to_lds(g0, g1, z4, z4, 0);
#endif
  }
  (void)buf;
#else
  stage_copy(buf, Wt, Kp, k0, n0, tid);
  (void)ldsoff;
#endif
}

// Complete previously issued staging (TDM path waits TENSORcnt; copy path relies on barrier's DS wait).
__device__ __forceinline__ void stage_fence(int tid) {
#if USE_TDM
  if (tid < 32) __builtin_amdgcn_s_wait_tensorcnt(0);
#else
  (void)tid;
#endif
}

// ---------------------------------------------------------------- zero fill
__global__ void __launch_bounds__(256) k_zero(float* __restrict__ p, size_t n) {
  size_t i = (size_t)blockIdx.x * 256 + threadIdx.x;
  size_t stride = (size_t)gridDim.x * 256;
  for (; i < n; i += stride) p[i] = 0.0f;
}

// --------------------------------- W[K][Nw] f32 -> Wt[Nw][Kp] bf16 (transpose, zero-pad K->Kp)
__global__ void __launch_bounds__(256) k_wt(const float* __restrict__ W, __bf16* __restrict__ Wt,
                                            int K, int Kp, int Nw, int total) {
  int i = blockIdx.x * 256 + threadIdx.x;
  if (i >= total) return;
  int n = i / Kp, k = i - n * Kp;
  Wt[i] = (__bf16)(k < K ? W[(size_t)k * Nw + n] : 0.0f);
}

// ------------------------------------------- h = relu(x @ node_w + node_b) -> bf16
__global__ void __launch_bounds__(256) k_node_proj(const float* __restrict__ x,
    const __bf16* __restrict__ Wt, const float* __restrict__ bias, __bf16* __restrict__ h) {
  extern __shared__ __bf16 bl[];
  const int tid = threadIdx.x, lane = tid & 31;
  const int rowtile = blockIdx.x * 8 + (tid >> 5);
  const int n0 = blockIdx.y * 64;
  const int T = NN / 16;
  const int rt = rowtile < T ? rowtile : T - 1;
  const float* xr = x + (size_t)(rt * 16 + (lane & 15)) * FN;
  v8f acc[4] = {};
  const int NCH = FN / 32;
  stage_issue(bl, Wt, FN, 0, n0, tid, 0u);
  for (int kc = 0; kc < NCH; ++kc) {
    stage_fence(tid);
    __syncthreads();
    if (kc + 1 < NCH)
      stage_issue(bl + ((kc + 1) & 1) * LDS_CHUNK, Wt, FN, (kc + 1) * 32, n0, tid,
                  (unsigned)(((kc + 1) & 1) * LDS_CHUNK * 2));
    const __bf16* cur = bl + (kc & 1) * LDS_CHUNK;
    const int run0 = kc * 32 + ((lane >> 4) << 3);
    v8f f0 = *(const v8f*)(xr + run0);
    v8f f1 = *(const v8f*)(xr + run0 + 16);
    if (kc + 1 < NCH) __builtin_prefetch(xr + run0 + 32, 0, 3);
    v16bf a;
    #pragma unroll
    for (int j = 0; j < 8; ++j) { a[j] = (__bf16)san(f0[j]); a[8 + j] = (__bf16)san(f1[j]); }
    #pragma unroll
    for (int s = 0; s < 4; ++s)
      acc[s] = __builtin_amdgcn_wmma_f32_16x16x32_bf16(false, a, false,
                 load_b_frag(cur, lane, s), (short)0, acc[s], false, false);
  }
  if (rowtile < T) {
    const int mhi = (lane >> 4) << 3, coli = lane & 15;
    #pragma unroll
    for (int s = 0; s < 4; ++s) {
      int col = n0 + s * 16 + coli;
      float bia = bias[col];
      #pragma unroll
      for (int i = 0; i < 8; ++i)
        h[(size_t)(rt * 16 + mhi + i) * HH + col] = (__bf16)fmaxf(acc[s][i] + bia, 0.0f);
    }
  }
}

// ------------------------------------------- e = relu(edge_attr @ edge_w + edge_b) -> bf16
__global__ void __launch_bounds__(256) k_edge_proj(const float* __restrict__ ea,
    const __bf16* __restrict__ Wt, const float* __restrict__ bias, __bf16* __restrict__ e) {
  extern __shared__ __bf16 bl[];
  const int tid = threadIdx.x, lane = tid & 31;
  const int rowtile = blockIdx.x * 8 + (tid >> 5);   // exact: EE/16/8 blocks
  const int n0 = blockIdx.y * 64;
  const float* ar = ea + (size_t)(rowtile * 16 + (lane & 15)) * FE;
  stage_issue(bl, Wt, 32, 0, n0, tid, 0u);   // Kp=32 (zero-padded beyond K=16), single chunk
  stage_fence(tid);
  __syncthreads();
  const int run0 = (lane >> 4) << 3;    // 0 or 8; K-run 16..31 is zero (B rows are zero there too)
  v8f f0 = *(const v8f*)(ar + run0);
  v16bf a;
  #pragma unroll
  for (int j = 0; j < 8; ++j) { a[j] = (__bf16)san(f0[j]); a[8 + j] = (__bf16)0.0f; }
  v8f acc[4] = {};
  #pragma unroll
  for (int s = 0; s < 4; ++s)
    acc[s] = __builtin_amdgcn_wmma_f32_16x16x32_bf16(false, a, false,
               load_b_frag(bl, lane, s), (short)0, acc[s], false, false);
  const int mhi = (lane >> 4) << 3, coli = lane & 15;
  #pragma unroll
  for (int s = 0; s < 4; ++s) {
    int col = n0 + s * 16 + coli;
    float bia = bias[col];
    #pragma unroll
    for (int i = 0; i < 8; ++i)
      e[(size_t)(rowtile * 16 + mhi + i) * HH + col] = (__bf16)fmaxf(acc[s][i] + bia, 0.0f);
  }
}

// ------------------------------------------- agg[dst] += relu(h[src] + e)   (f32 atomics)
__global__ void __launch_bounds__(256) k_msg(const __bf16* __restrict__ h,
    const __bf16* __restrict__ e, const int* __restrict__ src, const int* __restrict__ dst,
    float* __restrict__ agg) {
  size_t idx = (size_t)blockIdx.x * 256 + threadIdx.x;   // EE*32 work items, exact grid
  int ei = (int)(idx >> 5);
  int c0 = ((int)idx & 31) * 8;
  int s = src[ei], d = dst[ei];
  v8bf hv = *(const v8bf*)(h + (size_t)s * HH + c0);
  v8bf ev = *(const v8bf*)(e + (size_t)ei * HH + c0);
  float* ap = agg + (size_t)d * HH + c0;
  #pragma unroll
  for (int i = 0; i < 8; ++i)
    atomicAdd(ap + i, fmaxf((float)hv[i] + (float)ev[i], 0.0f));
}

// ------------------------------------------- t = relu((h+agg) @ w1 + b1) -> bf16 [N,512]
__global__ void __launch_bounds__(256) k_gemm1(const __bf16* __restrict__ h,
    const float* __restrict__ agg, const __bf16* __restrict__ Wt, const float* __restrict__ bias,
    __bf16* __restrict__ t) {
  extern __shared__ __bf16 bl[];
  const int tid = threadIdx.x, lane = tid & 31;
  const int rowtile = blockIdx.x * 8 + (tid >> 5);
  const int n0 = blockIdx.y * 64;
  const int T = NN / 16;
  const int rt = rowtile < T ? rowtile : T - 1;
  const __bf16* hr = h + (size_t)(rt * 16 + (lane & 15)) * HH;
  const float*  gr = agg + (size_t)(rt * 16 + (lane & 15)) * HH;
  v8f acc[4] = {};
  const int NCH = HH / 32;
  stage_issue(bl, Wt, HH, 0, n0, tid, 0u);
  for (int kc = 0; kc < NCH; ++kc) {
    stage_fence(tid);
    __syncthreads();
    if (kc + 1 < NCH)
      stage_issue(bl + ((kc + 1) & 1) * LDS_CHUNK, Wt, HH, (kc + 1) * 32, n0, tid,
                  (unsigned)(((kc + 1) & 1) * LDS_CHUNK * 2));
    const __bf16* cur = bl + (kc & 1) * LDS_CHUNK;
    const int run0 = kc * 32 + ((lane >> 4) << 3);
    v8bf h0 = *(const v8bf*)(hr + run0);
    v8bf h1 = *(const v8bf*)(hr + run0 + 16);
    v8f  g0 = *(const v8f*)(gr + run0);
    v8f  g1 = *(const v8f*)(gr + run0 + 16);
    if (kc + 1 < NCH) __builtin_prefetch(gr + run0 + 32, 0, 3);
    v16bf a;
    #pragma unroll
    for (int j = 0; j < 8; ++j) {
      a[j]     = (__bf16)((float)h0[j] + g0[j]);
      a[8 + j] = (__bf16)((float)h1[j] + g1[j]);
    }
    #pragma unroll
    for (int s = 0; s < 4; ++s)
      acc[s] = __builtin_amdgcn_wmma_f32_16x16x32_bf16(false, a, false,
                 load_b_frag(cur, lane, s), (short)0, acc[s], false, false);
  }
  if (rowtile < T) {
    const int mhi = (lane >> 4) << 3, coli = lane & 15;
    #pragma unroll
    for (int s = 0; s < 4; ++s) {
      int col = n0 + s * 16 + coli;
      float bia = bias[col];
      #pragma unroll
      for (int i = 0; i < 8; ++i)
        t[(size_t)(rt * 16 + mhi + i) * H2 + col] = (__bf16)fmaxf(acc[s][i] + bia, 0.0f);
    }
  }
}

// ------------------------------------------- zz = t @ w2 + b2 (f32) + BN column sums
__global__ void __launch_bounds__(256) k_gemm2(const __bf16* __restrict__ t,
    const __bf16* __restrict__ Wt, const float* __restrict__ bias, float* __restrict__ zz,
    float* __restrict__ cs, float* __restrict__ csq) {
  extern __shared__ __bf16 bl[];
  const int tid = threadIdx.x, lane = tid & 31;
  const int rowtile = blockIdx.x * 8 + (tid >> 5);
  const int n0 = blockIdx.y * 64;
  const int T = NN / 16;
  const int rt = rowtile < T ? rowtile : T - 1;
  const __bf16* tr = t + (size_t)(rt * 16 + (lane & 15)) * H2;
  v8f acc[4] = {};
  const int NCH = H2 / 32;
  stage_issue(bl, Wt, H2, 0, n0, tid, 0u);
  for (int kc = 0; kc < NCH; ++kc) {
    stage_fence(tid);
    __syncthreads();
    if (kc + 1 < NCH)
      stage_issue(bl + ((kc + 1) & 1) * LDS_CHUNK, Wt, H2, (kc + 1) * 32, n0, tid,
                  (unsigned)(((kc + 1) & 1) * LDS_CHUNK * 2));
    const __bf16* cur = bl + (kc & 1) * LDS_CHUNK;
    const int run0 = kc * 32 + ((lane >> 4) << 3);
    v8bf t0 = *(const v8bf*)(tr + run0);
    v8bf t1 = *(const v8bf*)(tr + run0 + 16);
    if (kc + 1 < NCH) __builtin_prefetch(tr + run0 + 32, 0, 3);
    v16bf a = __builtin_shufflevector(t0, t1, 0, 1, 2, 3, 4, 5, 6, 7, 8, 9, 10, 11, 12, 13, 14, 15);
    #pragma unroll
    for (int s = 0; s < 4; ++s)
      acc[s] = __builtin_amdgcn_wmma_f32_16x16x32_bf16(false, a, false,
                 load_b_frag(cur, lane, s), (short)0, acc[s], false, false);
  }
  if (rowtile < T) {
    const int mhi = (lane >> 4) << 3, coli = lane & 15;
    #pragma unroll
    for (int s = 0; s < 4; ++s) {
      int col = n0 + s * 16 + coli;
      float bia = bias[col];
      float s1 = 0.0f, s2 = 0.0f;
      #pragma unroll
      for (int i = 0; i < 8; ++i) {
        float v = acc[s][i] + bia;
        zz[(size_t)(rt * 16 + mhi + i) * HH + col] = v;
        s1 += v; s2 += v * v;
      }
      atomicAdd(&cs[col],  s1);
      atomicAdd(&csq[col], s2);
    }
  }
}

// ------------------------------------------- h = relu(BN(zz)) -> bf16
__global__ void __launch_bounds__(256) k_bn(const float* __restrict__ zz,
    const float* __restrict__ cs, const float* __restrict__ csq,
    const float* __restrict__ gamma, const float* __restrict__ beta, __bf16* __restrict__ h) {
  size_t i = (size_t)blockIdx.x * 256 + threadIdx.x;   // grid exactly NN*HH/256
  int c = (int)(i & (HH - 1));
  float mean = cs[c] * (1.0f / NN);
  float var  = csq[c] * (1.0f / NN) - mean * mean;
  float xh = (zz[i] - mean) * rsqrtf(var + 1e-5f);
  h[i] = (__bf16)fmaxf(gamma[c] * xh + beta[c], 0.0f);
}

// ------------------------------------------- gate scores + segment max (one wave per node)
__global__ void __launch_bounds__(256) k_gate(const __bf16* __restrict__ h,
    const float* __restrict__ gw, const float* __restrict__ gb, const int* __restrict__ batch,
    float* __restrict__ g, unsigned* __restrict__ gmax) {
  int node = blockIdx.x * 8 + (threadIdx.x >> 5);   // exact: NN/8 blocks
  int lane = threadIdx.x & 31;
  const __bf16* hp = h + (size_t)node * HH + lane * 8;
  float s = 0.0f;
  #pragma unroll
  for (int i = 0; i < 8; ++i) s += (float)hp[i] * gw[lane * 8 + i];
  #pragma unroll
  for (int o = 16; o > 0; o >>= 1) s += __shfl_xor(s, o, 32);
  if (lane == 0) {
    float gg = s + gb[0];
    g[node] = gg;
    unsigned b = __float_as_uint(gg);
    unsigned enc = (b & 0x80000000u) ? ~b : (b | 0x80000000u);  // order-preserving encode
    atomicMax(&gmax[batch[node]], enc);
  }
}

// ------------------------------------------- ex = exp(g - gmax), denom = segment sum
__global__ void __launch_bounds__(256) k_softmax_num(const float* __restrict__ g,
    const unsigned* __restrict__ gmax, const int* __restrict__ batch,
    float* __restrict__ ex, float* __restrict__ denom) {
  int i = blockIdx.x * 256 + threadIdx.x;
  if (i >= NN) return;
  int bg = batch[i];
  unsigned u = gmax[bg];
  unsigned b = (u & 0x80000000u) ? (u & 0x7fffffffu) : ~u;
  float e = __expf(g[i] - __uint_as_float(b));
  ex[i] = e;
  atomicAdd(&denom[bg], e);
}

// ------------------------------------------- out[batch] += alpha * (h @ pool_w + pool_b)
__global__ void __launch_bounds__(256) k_pool(const __bf16* __restrict__ h,
    const __bf16* __restrict__ Wt, const float* __restrict__ bias, const float* __restrict__ ex,
    const float* __restrict__ denom, const int* __restrict__ batch, float* __restrict__ out) {
  extern __shared__ __bf16 bl[];
  const int tid = threadIdx.x, lane = tid & 31;
  const int rowtile = blockIdx.x * 8 + (tid >> 5);
  const int n0 = blockIdx.y * 64;
  const int T = NN / 16;
  const int rt = rowtile < T ? rowtile : T - 1;
  const __bf16* hr = h + (size_t)(rt * 16 + (lane & 15)) * HH;
  v8f acc[4] = {};
  const int NCH = HH / 32;
  stage_issue(bl, Wt, HH, 0, n0, tid, 0u);
  for (int kc = 0; kc < NCH; ++kc) {
    stage_fence(tid);
    __syncthreads();
    if (kc + 1 < NCH)
      stage_issue(bl + ((kc + 1) & 1) * LDS_CHUNK, Wt, HH, (kc + 1) * 32, n0, tid,
                  (unsigned)(((kc + 1) & 1) * LDS_CHUNK * 2));
    const __bf16* cur = bl + (kc & 1) * LDS_CHUNK;
    const int run0 = kc * 32 + ((lane >> 4) << 3);
    v8bf h0 = *(const v8bf*)(hr + run0);
    v8bf h1 = *(const v8bf*)(hr + run0 + 16);
    v16bf a = __builtin_shufflevector(h0, h1, 0, 1, 2, 3, 4, 5, 6, 7, 8, 9, 10, 11, 12, 13, 14, 15);
    #pragma unroll
    for (int s = 0; s < 4; ++s)
      acc[s] = __builtin_amdgcn_wmma_f32_16x16x32_bf16(false, a, false,
                 load_b_frag(cur, lane, s), (short)0, acc[s], false, false);
  }
  if (rowtile < T) {
    const int mhi = (lane >> 4) << 3, coli = lane & 15;
    #pragma unroll
    for (int s = 0; s < 4; ++s) {
      int col = n0 + s * 16 + coli;
      float bia = bias[col];
      #pragma unroll
      for (int i = 0; i < 8; ++i) {
        int r = rt * 16 + mhi + i;
        int bg = batch[r];
        float alpha = ex[r] / denom[bg];
        atomicAdd(&out[(size_t)bg * HH + col], alpha * (acc[s][i] + bia));
      }
    }
  }
}

// ============================================================== host launcher
extern "C" void kernel_launch(void* const* d_in, const int* in_sizes, int n_in,
                              void* d_out, int out_size, void* d_ws, size_t ws_size,
                              hipStream_t stream) {
  const float* x      = (const float*)d_in[0];
  const float* eattr  = (const float*)d_in[1];
  const int*   eidx   = (const int*)d_in[2];
  const int*   batch  = (const int*)d_in[3];
  const float* node_w = (const float*)d_in[4];
  const float* node_b = (const float*)d_in[5];
  const float* edge_w = (const float*)d_in[6];
  const float* edge_b = (const float*)d_in[7];
  const float* w1     = (const float*)d_in[8];
  const float* b1     = (const float*)d_in[9];
  const float* w2     = (const float*)d_in[10];
  const float* b2     = (const float*)d_in[11];
  const float* gamma  = (const float*)d_in[12];
  const float* beta   = (const float*)d_in[13];
  const float* gate_w = (const float*)d_in[14];
  const float* gate_b = (const float*)d_in[15];
  const float* pool_w = (const float*)d_in[16];
  const float* pool_b = (const float*)d_in[17];
  const int* src = eidx;
  const int* dst = eidx + EE;
  float* out = (float*)d_out;

  char* p = (char*)d_ws;
  size_t off = 0;
  auto take = [&](size_t bytes) { char* r = p + off; off += (bytes + 255) & ~(size_t)255; return r; };
  __bf16* e_bf   = (__bf16*)take((size_t)EE * HH * 2);
  __bf16* h_bf   = (__bf16*)take((size_t)NN * HH * 2);
  float*  agg    = (float*)take(((size_t)NN * HH + 2 * HH) * 4);  // agg | colsum | colsumsq
  float*  cs     = agg + (size_t)NN * HH;
  float*  csq    = cs + HH;
  __bf16* t_bf   = (__bf16*)take((size_t)NN * H2 * 2);
  float*  zz     = (float*)take((size_t)NN * HH * 4);
  float*  gbuf   = (float*)take((size_t)NN * 4);
  float*  exbf   = (float*)take((size_t)NN * 4);
  float*  sm     = (float*)take((size_t)(GG * 2) * 4);            // gmax(enc) | denom
  unsigned* gmax = (unsigned*)sm;
  float*  denom  = sm + GG;
  // pre-transposed bf16 weights: Wt[n][Kp]
  __bf16* wt_node = (__bf16*)take((size_t)HH * FN * 2);          // 256 x 128
  __bf16* wt_edge = (__bf16*)take((size_t)HH * 32 * 2);          // 256 x 32 (K=16 padded)
  __bf16* wt1     = (__bf16*)take((size_t)LL * H2 * HH * 2);     // per layer: 512 x 256
  __bf16* wt2     = (__bf16*)take((size_t)LL * HH * H2 * 2);     // per layer: 256 x 512
  __bf16* wt_pool = (__bf16*)take((size_t)HH * HH * 2);          // 256 x 256

  dim3 blk(256);
  // weight convert + transpose (bf16, K padded to multiple of 32)
  k_wt<<<dim3((HH * FN + 255) / 256), blk, 0, stream>>>(node_w, wt_node, FN, FN, HH, HH * FN);
  k_wt<<<dim3((HH * 32 + 255) / 256), blk, 0, stream>>>(edge_w, wt_edge, FE, 32, HH, HH * 32);
  for (int l = 0; l < LL; ++l) {
    k_wt<<<dim3((H2 * HH + 255) / 256), blk, 0, stream>>>(w1 + (size_t)l * HH * H2,
        wt1 + (size_t)l * H2 * HH, HH, HH, H2, H2 * HH);
    k_wt<<<dim3((HH * H2 + 255) / 256), blk, 0, stream>>>(w2 + (size_t)l * H2 * HH,
        wt2 + (size_t)l * HH * H2, H2, H2, HH, HH * H2);
  }
  k_wt<<<dim3((HH * HH + 255) / 256), blk, 0, stream>>>(pool_w, wt_pool, HH, HH, HH, HH * HH);

  k_node_proj<<<dim3(391, 4), blk, LDS_BYTES, stream>>>(x, wt_node, node_b, h_bf);
  k_edge_proj<<<dim3(6250, 4), blk, LDS_BYTES, stream>>>(eattr, wt_edge, edge_b, e_bf);
  for (int l = 0; l < LL; ++l) {
    k_zero<<<dim3(2048), blk, 0, stream>>>(agg, (size_t)NN * HH + 2 * HH);
    k_msg<<<dim3(100000), blk, 0, stream>>>(h_bf, e_bf, src, dst, agg);
    k_gemm1<<<dim3(391, 8), blk, LDS_BYTES, stream>>>(h_bf, agg, wt1 + (size_t)l * H2 * HH,
                                                      b1 + l * H2, t_bf);
    k_gemm2<<<dim3(391, 4), blk, LDS_BYTES, stream>>>(t_bf, wt2 + (size_t)l * HH * H2,
                                                      b2 + l * HH, zz, cs, csq);
    k_bn<<<dim3(50000), blk, 0, stream>>>(zz, cs, csq, gamma + l * HH, beta + l * HH, h_bf);
  }
  k_zero<<<dim3(16), blk, 0, stream>>>(out, (size_t)GG * HH);
  k_zero<<<dim3(1), blk, 0, stream>>>(sm, (size_t)GG * 2);
  k_gate<<<dim3(6250), blk, 0, stream>>>(h_bf, gate_w, gate_b, batch, gbuf, gmax);
  k_softmax_num<<<dim3((NN + 255) / 256), blk, 0, stream>>>(gbuf, gmax, batch, exbf, denom);
  k_pool<<<dim3(391, 4), blk, LDS_BYTES, stream>>>(h_bf, wt_pool, pool_b, exbf, denom, batch, out);
  (void)in_sizes; (void)n_in; (void)out_size; (void)ws_size;
}
